// Ultra_858993459634
// MI455X (gfx1250) — compile-verified, matrix operand order checked by reading, not run on hardware
//
#include <hip/hip_runtime.h>
#include <hip/hip_bf16.h>

#define Bq 4
#define Nn 20000
#define Ee 320000
#define Dd 64
#define Rr 64
#define Tt 1000
#define Ll 6
#define LN_EPS 1e-5f

#define WT_STRIDE 136   // padded k-major row stride (halves); 272B = 17*16B
#define OUT_STRIDE 65   // padded f32 row stride for LN tile

typedef __attribute__((ext_vector_type(16))) _Float16 v16h;
typedef __attribute__((ext_vector_type(8)))  _Float16 v8h;
typedef __attribute__((ext_vector_type(8)))  float    v8f;

__device__ __forceinline__ void atomAddF(float* p, float v) {
#if defined(__AMDGCN__)
    unsafeAtomicAdd(p, v);   // hw global_atomic_add_f32 (RMW resolved in L2)
#else
    atomicAdd(p, v);
#endif
}

// Load 16 halves of an A fragment: 8 consecutive f32 at p[0..7] (j=0..7)
// and 8 consecutive f32 at p[16..23] (j=8..15), per ISA 16-bit A layout.
__device__ __forceinline__ v16h loadA16(const float* __restrict__ p) {
    const float4 a = *(const float4*)(p);
    const float4 b = *(const float4*)(p + 4);
    const float4 c = *(const float4*)(p + 16);
    const float4 d = *(const float4*)(p + 20);
    v16h r;
    r[0]  = (_Float16)a.x; r[1]  = (_Float16)a.y; r[2]  = (_Float16)a.z; r[3]  = (_Float16)a.w;
    r[4]  = (_Float16)b.x; r[5]  = (_Float16)b.y; r[6]  = (_Float16)b.z; r[7]  = (_Float16)b.w;
    r[8]  = (_Float16)c.x; r[9]  = (_Float16)c.y; r[10] = (_Float16)c.z; r[11] = (_Float16)c.w;
    r[12] = (_Float16)d.x; r[13] = (_Float16)d.y; r[14] = (_Float16)d.z; r[15] = (_Float16)d.w;
    return r;
}

// ---------------------------------------------------------------------------
// query[b] = rel_reps[b, r_index[b]]; scatter into x0[b, h_index[b]] (x0 zeroed).
// ---------------------------------------------------------------------------
__global__ void boundary_kernel(const float* __restrict__ relreps,
                                const int* __restrict__ r_index,
                                const int* __restrict__ h_index,
                                float* __restrict__ x0,
                                float* __restrict__ query) {
    int tid = threadIdx.x;               // 0..255
    int b = tid >> 6, d = tid & 63;
    float q = relreps[(size_t)b * Rr * Dd + (size_t)r_index[b] * Dd + d];
    query[tid] = q;
    x0[(size_t)b * Nn * Dd + (size_t)h_index[b] * Dd + d] = q;
}

// agg[b, h_index[b], :] += query[b, :]   (boundary self-loop, hoisted out of GEMM)
__global__ void add_boundary_kernel(const float* __restrict__ query,
                                    const int* __restrict__ h_index,
                                    float* __restrict__ agg) {
    int tid = threadIdx.x;               // 0..255
    int b = tid >> 6, d = tid & 63;
    agg[(size_t)b * Nn * Dd + (size_t)h_index[b] * Dd + d] += query[tid];
}

// ---------------------------------------------------------------------------
// Relation projection MLP: rel[b,r,:] = relu(rr @ w1 + b1) @ w2 + b2
// ---------------------------------------------------------------------------
__global__ void relproj_kernel(const float* __restrict__ relreps,
                               const float* __restrict__ w1,
                               const float* __restrict__ b1,
                               const float* __restrict__ w2,
                               const float* __restrict__ b2,
                               float* __restrict__ rel) {
    __shared__ float rr[Rr * Dd];
    __shared__ float hid[Rr * Dd];
    int b = blockIdx.x, tid = threadIdx.x;
    const float* src = relreps + (size_t)b * Rr * Dd;
    for (int i = tid; i < Rr * Dd; i += blockDim.x) rr[i] = src[i];
    __syncthreads();
    for (int i = tid; i < Rr * Dd; i += blockDim.x) {
        int r = i >> 6, d = i & 63;
        float acc = b1[d];
        #pragma unroll 8
        for (int k = 0; k < Dd; ++k) acc += rr[r * Dd + k] * w1[k * Dd + d];
        hid[i] = fmaxf(acc, 0.f);
    }
    __syncthreads();
    for (int i = tid; i < Rr * Dd; i += blockDim.x) {
        int r = i >> 6, d = i & 63;
        float acc = b2[d];
        #pragma unroll 8
        for (int k = 0; k < Dd; ++k) acc += hid[r * Dd + k] * w2[k * Dd + d];
        rel[(size_t)b * Rr * Dd + i] = acc;
    }
}

// ---------------------------------------------------------------------------
// DistMult message + scatter-add: agg[b,dst] += x[b,src] * rel[b,et] * w
// thread = (edge, dim); x/agg/rel stay L2-resident (192 MB L2).
// ---------------------------------------------------------------------------
__global__ void msg_kernel(const int* __restrict__ src,
                           const int* __restrict__ dst,
                           const int* __restrict__ etype,
                           const float* __restrict__ ew,
                           const float* __restrict__ x,
                           const float* __restrict__ rel,
                           float* __restrict__ agg) {
    long long idx = (long long)blockIdx.x * blockDim.x + threadIdx.x;
    int e = (int)(idx >> 6), d = (int)(idx & 63);
    if (e >= Ee) return;
    int s = src[e], t = dst[e], r = etype[e];
    float w = ew[e];
    #pragma unroll
    for (int b = 0; b < Bq; ++b) {
        float v = x[(size_t)b * Nn * Dd + (size_t)s * Dd + d] *
                  rel[(size_t)b * Rr * Dd + (size_t)r * Dd + d] * w;
        atomAddF(&agg[(size_t)b * Nn * Dd + (size_t)t * Dd + d], v);
    }
}

// ---------------------------------------------------------------------------
// Update: h = LN(concat([x, agg]) @ cw + cb); out = relu(h) + x
// 8 waves/block; wave owns a 16-row M-tile x 64 cols; K=128 in 4 chunks.
// Weights in LDS k-major (transposed, padded) so each B fragment is two
// 16-byte ds_load_b128 per lane. LN epilogue on padded LDS tile.
// ---------------------------------------------------------------------------
__global__ __launch_bounds__(256) void conv_ln_kernel(
        const float* __restrict__ xin,
        const float* __restrict__ agg,
        const float* __restrict__ cw,   // [128,64] (k-major in global)
        const float* __restrict__ cb,   // [64]
        const float* __restrict__ lng,
        const float* __restrict__ lnb,
        float* __restrict__ xout) {
    __shared__ _Float16 wT[64 * WT_STRIDE];     // transposed weights [n][k], ~17.4 KB
    __shared__ float    outLds[128 * OUT_STRIDE]; // 8 waves x 16x64 tile, padded

    int tid = threadIdx.x;
    // stage weights: coalesced global read, transpose into k-major LDS rows
    for (int i = tid; i < 128 * 64; i += 256) {
        int k = i >> 6, n = i & 63;
        wT[n * WT_STRIDE + k] = (_Float16)cw[i];
    }
    __syncthreads();

    int wave = tid >> 5;
    int lane = tid & 31;
    int mloc = lane & 15;     // M (A) or N (B, C/D) within 16
    int grp  = lane >> 4;

    int row = blockIdx.x * 128 + wave * 16 + mloc;  // tiles never cross batch (N%16==0)
    const float* xrow = xin + (size_t)row * Dd;
    const float* arow = agg + (size_t)row * Dd;

    v8f acc0 = {}, acc1 = {}, acc2 = {}, acc3 = {};

    #pragma unroll
    for (int kc = 0; kc < 4; ++kc) {
        // ---- A fragment: k in [kc*32, kc*32+32); first half from x, second from agg
        const float* srcp = (kc < 2) ? xrow : arow;
        int kOff = (kc & 1) * 32 + (grp << 3);
        v16h afrag = loadA16(srcp + kOff);

        // ---- B fragments: two ds_load_b128 per n-tile from k-major LDS rows
        #pragma unroll
        for (int t = 0; t < 4; ++t) {
            const _Float16* bp = &wT[(t * 16 + mloc) * WT_STRIDE + kc * 32 + (grp << 3)];
            v8h lo = *(const v8h*)bp;          // k0 .. k0+7   (j = 0..7)
            v8h hi = *(const v8h*)(bp + 16);   // k0+16 .. +23 (j = 8..15)
            v16h bfrag = __builtin_shufflevector(lo, hi,
                0, 1, 2, 3, 4, 5, 6, 7, 8, 9, 10, 11, 12, 13, 14, 15);
            v8f* accp = (t == 0) ? &acc0 : (t == 1) ? &acc1 : (t == 2) ? &acc2 : &acc3;
            *accp = __builtin_amdgcn_wmma_f32_16x16x32_f16(
                false, afrag, false, bfrag, (short)0, *accp, false, false);
        }
    }

    // ---- spill C/D tile (+bias): lane=N, elem v -> M = v + 8*grp ----
    #pragma unroll
    for (int t = 0; t < 4; ++t) {
        v8f a = (t == 0) ? acc0 : (t == 1) ? acc1 : (t == 2) ? acc2 : acc3;
        int n = t * 16 + mloc;
        float bias = cb[n];
        #pragma unroll
        for (int v = 0; v < 8; ++v) {
            int m = v + (grp << 3);
            outLds[(wave * 16 + m) * OUT_STRIDE + n] = a[v] + bias;
        }
    }
    __syncthreads();

    // ---- LayerNorm + relu + residual: one thread per row, conflict-free reads
    if (tid < 128) {
        int r = blockIdx.x * 128 + tid;
        const float* orow = &outLds[tid * OUT_STRIDE];
        float mu = 0.f;
        #pragma unroll 8
        for (int c = 0; c < 64; ++c) mu += orow[c];
        mu *= (1.f / 64.f);
        float var = 0.f;
        #pragma unroll 8
        for (int c = 0; c < 64; ++c) { float d = orow[c] - mu; var += d * d; }
        var *= (1.f / 64.f);
        float rstd = rsqrtf(var + LN_EPS);
        const float* xr = xin + (size_t)r * Dd;
        float* xo = xout + (size_t)r * Dd;
        #pragma unroll 4
        for (int c = 0; c < 64; ++c) {
            float v = (orow[c] - mu) * rstd * lng[c] + lnb[c];
            xo[c] = fmaxf(v, 0.f) + xr[c];
        }
    }
}

// ---------------------------------------------------------------------------
// Final scoring MLP: one block (128 thr) per (b,t) row.
// ---------------------------------------------------------------------------
__global__ void score_kernel(const float* __restrict__ x,
                             const float* __restrict__ query,
                             const int* __restrict__ t_index,
                             const float* __restrict__ w1,  // [128,128]
                             const float* __restrict__ b1,  // [128]
                             const float* __restrict__ w2,  // [128,1]
                             const float* __restrict__ b2,  // [1]
                             float* __restrict__ out) {
    __shared__ float feat[128];
    __shared__ float hred[128];
    int row = blockIdx.x;            // 0 .. B*T-1
    int b = row / Tt;
    int tid = threadIdx.x;
    int node = t_index[row];
    feat[tid] = (tid < 64) ? x[(size_t)b * Nn * Dd + (size_t)node * Dd + tid]
                           : query[b * Dd + (tid - 64)];
    __syncthreads();
    float acc = b1[tid];
    #pragma unroll 8
    for (int k = 0; k < 128; ++k) acc += feat[k] * w1[k * 128 + tid];
    hred[tid] = fmaxf(acc, 0.f) * w2[tid];
    __syncthreads();
    #pragma unroll
    for (int s = 64; s > 0; s >>= 1) {
        if (tid < s) hred[tid] += hred[tid + s];
        __syncthreads();
    }
    if (tid == 0) out[row] = hred[0] + b2[0];
}

// ---------------------------------------------------------------------------
extern "C" void kernel_launch(void* const* d_in, const int* in_sizes, int n_in,
                              void* d_out, int out_size, void* d_ws, size_t ws_size,
                              hipStream_t stream) {
    const int*   edge_index = (const int*)d_in[0];     // [2,E]
    const int*   edge_type  = (const int*)d_in[1];     // [E]
    const float* relreps    = (const float*)d_in[2];   // [B,R,D]
    const int*   h_index    = (const int*)d_in[3];     // [B]
    const int*   r_index    = (const int*)d_in[4];     // [B]
    const int*   t_index    = (const int*)d_in[5];     // [B,T]
    const float* edge_w     = (const float*)d_in[6];   // [E]
    const float* rp_w1      = (const float*)d_in[7];   // [L,D,D]
    const float* rp_b1      = (const float*)d_in[8];   // [L,D]
    const float* rp_w2      = (const float*)d_in[9];   // [L,D,D]
    const float* rp_b2      = (const float*)d_in[10];  // [L,D]
    const float* conv_w     = (const float*)d_in[11];  // [L,2D,D]
    const float* conv_b     = (const float*)d_in[12];  // [L,D]
    const float* ln_g       = (const float*)d_in[13];  // [L,D]
    const float* ln_b       = (const float*)d_in[14];  // [L,D]
    const float* mlp_w1     = (const float*)d_in[15];  // [2D,2D]
    const float* mlp_b1     = (const float*)d_in[16];  // [2D]
    const float* mlp_w2     = (const float*)d_in[17];  // [2D,1]
    const float* mlp_b2     = (const float*)d_in[18];  // [1]

    const size_t xElems = (size_t)Bq * Nn * Dd;        // 5,120,000 floats
    float* x0    = (float*)d_ws;
    float* x1    = x0 + xElems;
    float* agg   = x1 + xElems;
    float* rel   = agg + xElems;
    float* query = rel + (size_t)Bq * Rr * Dd;

    const int* e_src = edge_index;
    const int* e_dst = edge_index + Ee;

    // boundary init
    hipMemsetAsync(x0, 0, xElems * sizeof(float), stream);
    boundary_kernel<<<1, 256, 0, stream>>>(relreps, r_index, h_index, x0, query);

    const int msgBlocks = (Ee * 64) / 256;             // 80000
    for (int l = 0; l < Ll; ++l) {
        const float* xin  = (l & 1) ? x1 : x0;
        float*       xout = (l & 1) ? x0 : x1;
        relproj_kernel<<<Bq, 256, 0, stream>>>(
            relreps, rp_w1 + (size_t)l * Dd * Dd, rp_b1 + (size_t)l * Dd,
            rp_w2 + (size_t)l * Dd * Dd, rp_b2 + (size_t)l * Dd, rel);
        hipMemsetAsync(agg, 0, xElems * sizeof(float), stream);
        msg_kernel<<<msgBlocks, 256, 0, stream>>>(
            e_src, e_dst, edge_type, edge_w, xin, rel, agg);
        add_boundary_kernel<<<1, 256, 0, stream>>>(query, h_index, agg);
        conv_ln_kernel<<<(Bq * Nn) / 128, 256, 0, stream>>>(
            xin, agg,
            conv_w + (size_t)l * 2 * Dd * Dd, conv_b + (size_t)l * Dd,
            ln_g + (size_t)l * Dd, ln_b + (size_t)l * Dd, xout);
    }
    // after 6 layers (even), final state is in x0
    score_kernel<<<Bq * Tt, 128, 0, stream>>>(
        x0, query, t_index, mlp_w1, mlp_b1, mlp_w2, mlp_b2, (float*)d_out);
}
